// TransformerEncoder_52639119179966
// MI455X (gfx1250) — matrix operand order repaired and checked by reading, not compiled, not run on hardware
//
#include <hip/hip_runtime.h>
#include <hip/hip_bf16.h>
#include <stdint.h>

typedef __attribute__((ext_vector_type(16))) __bf16 v16bf;
typedef __attribute__((ext_vector_type(8)))  float  v8f;
typedef int v4i __attribute__((vector_size(16)));
typedef unsigned short u16;

#define D_MODEL 1024
#define N_HEAD  16
#define DHEAD   64
#define F_FF    4096
#define N_LAYER 6
#define LN_EPS  1e-12f

// ---- CDNA5 async global->LDS copy (ASYNCcnt path), guarded by __has_builtin ----
#if defined(__has_builtin)
#if __has_builtin(__builtin_amdgcn_global_load_async_to_lds_b128)
#define HAVE_ASYNC_LDS 1
#endif
#if __has_builtin(__builtin_amdgcn_s_wait_asynccnt)
#define HAVE_WAIT_ASYNC_BUILTIN 1
#endif
#if __has_builtin(__builtin_amdgcn_update_dpp)
#define HAVE_DPP 1
#endif
#endif
#ifndef HAVE_ASYNC_LDS
#define HAVE_ASYNC_LDS 0
#endif
#ifndef HAVE_WAIT_ASYNC_BUILTIN
#define HAVE_WAIT_ASYNC_BUILTIN 0
#endif
#ifndef HAVE_DPP
#define HAVE_DPP 0
#endif

#if HAVE_ASYNC_LDS
typedef __attribute__((address_space(1))) v4i gv4i;
typedef __attribute__((address_space(3))) v4i lv4i;
#endif

__device__ __forceinline__ void async_copy16(const u16* g, u16* l) {
#if HAVE_ASYNC_LDS
  // generic LDS pointer low 32 bits == LDS offset (ISA 10.2 aperture rules);
  // generic global pointer == AS1 pointer value.
  __builtin_amdgcn_global_load_async_to_lds_b128(
      (gv4i*)(uintptr_t)g,
      (lv4i*)(uint32_t)(uintptr_t)l,
      0, 0);
#else
  *(float4*)l = *(const float4*)g;
#endif
}

#if HAVE_WAIT_ASYNC_BUILTIN
#define WAIT_ASYNC(n) __builtin_amdgcn_s_wait_asynccnt(n)
#elif HAVE_ASYNC_LDS
#define WAIT_ASYNC(n) asm volatile("s_wait_asynccnt %0" ::"i"(n) : "memory")
#else
#define WAIT_ASYNC(n) ((void)0)
#endif

// XOR-butterfly max over lanes within a row of 16: DPP ROW_XMASK (0x160|mask),
// no LDS traffic; fallback to __shfl_xor (ds_bpermute).
#if HAVE_DPP
#define XORMAX(x, m) fmaxf((x), __int_as_float(__builtin_amdgcn_update_dpp( \
    0, __float_as_int(x), 0x160 + (m), 0xf, 0xf, true)))
#else
#define XORMAX(x, m) fmaxf((x), __shfl_xor((x), (m), 32))
#endif

// 16 bf16 fragment viewed as two 16B chunks or 16 scalars
union Frag16 {
  v16bf  v;
  float4 q[2];
  u16    s[16];
};

union F4U16 {
  float4 f;
  u16    s[8];
};

__device__ __forceinline__ u16 f2bf(float f) {
  uint32_t u = __float_as_uint(f);
  u += 0x7fffu + ((u >> 16) & 1u);   // round-to-nearest-even
  return (u16)(u >> 16);
}

__device__ __forceinline__ float gelu_exact(float x) {
  return 0.5f * x * (1.0f + erff(x * 0.70710678118654752f));
}

// ---------------- elementwise f32 -> bf16 ----------------
__global__ __launch_bounds__(256)
void cvt_f32_bf16(const float* __restrict__ in, u16* __restrict__ out, int n) {
  const int i = blockIdx.x * 256 + threadIdx.x;
  if (i < n) out[i] = f2bf(in[i]);
}

// ---------------- weight f32 (K,N) -> bf16 transposed (N,K) ----------------
__global__ __launch_bounds__(256)
void wcvt_t(const float* __restrict__ W, u16* __restrict__ WT, int K, int N) {
  const size_t i = (size_t)blockIdx.x * 256 + threadIdx.x;
  if (i >= (size_t)K * N) return;
  const int k = (int)(i / N);
  const int n = (int)(i % N);
  WT[(size_t)n * K + k] = f2bf(W[i]);
}

// ---------------- bf16 GEMM: C[M,N] = A[M,K] @ W[K,N] + bias, WT is (N,K) ----
// Block: 256 thr = 8 waves; block tile 128x128; wave tile 32x64 (2x4 wmma tiles).
// Double-buffered LDS, async global->LDS prefetch of next K tile.
// EPI: 0 = bias only, 1 = bias + exact GELU.
template <int EPI>
__global__ __launch_bounds__(256)
void gemm_bf16(const u16* __restrict__ A, const u16* __restrict__ WT,
               const float* __restrict__ bias,
               float* __restrict__ outf, u16* __restrict__ outb,
               int M, int N, int K) {
  __shared__ u16 As[2][128 * 40];  // 128 rows x 32 k, stride 40 (80B rows, 16B aligned)
  __shared__ u16 Bs[2][128 * 40];  // 128 n-rows x 32 k
  const int lane = threadIdx.x & 31;
  const int wave = threadIdx.x >> 5;
  const int wm   = wave & 3;          // 4 waves in M (32 rows each)
  const int wn   = wave >> 2;         // 2 waves in N (64 cols each)
  const int l16  = lane & 15;
  const int hl   = lane >> 4;
  const int m0   = blockIdx.x * 128;
  const int n0   = blockIdx.y * 128;
  v8f acc[2][4] = {};

  auto stage = [&](int kk, int bufi) {
#pragma unroll
    for (int i = 0; i < 2; ++i) {   // 512 16B chunks, 2 per thread per matrix
      const int idx = threadIdx.x + i * 256;
      const int row = idx >> 2;
      const int cg  = (idx & 3) * 8;
      async_copy16(A + (size_t)(m0 + row) * K + kk + cg, &As[bufi][row * 40 + cg]);
      async_copy16(WT + (size_t)(n0 + row) * K + kk + cg, &Bs[bufi][row * 40 + cg]);
    }
  };

  stage(0, 0);                 // 4 async ops outstanding
  int nbuf = 0;
  for (int k0 = 0; k0 < K; k0 += 32) {
    const bool more = (k0 + 32 < K);
    if (more) {
      stage(k0 + 32, nbuf ^ 1);  // +4 outstanding
      WAIT_ASYNC(4);             // current tile's 4 complete (in-order)
    } else {
      WAIT_ASYNC(0);
    }
    __syncthreads();             // all waves' current tile visible in LDS
    const u16* as = As[nbuf];
    const u16* bs = Bs[nbuf];
    // A fragments (16x32): lane l16 = row, half selects K octet (+16)
    Frag16 af[2];
#pragma unroll
    for (int s = 0; s < 2; ++s) {
      const int ar = wm * 32 + s * 16 + l16;
      af[s].q[0] = *(const float4*)(&as[ar * 40 + hl * 8]);
      af[s].q[1] = *(const float4*)(&as[ar * 40 + hl * 8 + 16]);
    }
    // B fragments (32x16): lane l16 = col, half selects K octet (+16)
#pragma unroll
    for (int t = 0; t < 4; ++t) {
      Frag16 bf;
      const int br = wn * 64 + t * 16 + l16;
      bf.q[0] = *(const float4*)(&bs[br * 40 + hl * 8]);
      bf.q[1] = *(const float4*)(&bs[br * 40 + hl * 8 + 16]);
#pragma unroll
      for (int s = 0; s < 2; ++s)
        acc[s][t] = __builtin_amdgcn_wmma_f32_16x16x32_bf16(false, af[s].v, false, bf.v,
                                                            (short)0, acc[s][t], false, false);
    }
    __syncthreads();             // compute done before buffer is overwritten
    nbuf ^= 1;
  }
  // C layout: VGPR r -> row r + 8*half, col = l16
#pragma unroll
  for (int t = 0; t < 4; ++t) {
    const int   ng  = n0 + wn * 64 + t * 16 + l16;
    const float bia = bias[ng];
#pragma unroll
    for (int s = 0; s < 2; ++s) {
#pragma unroll
      for (int r = 0; r < 8; ++r) {
        const int mg = m0 + wm * 32 + s * 16 + r + 8 * hl;
        float v0 = acc[s][t][r] + bia;
        if (EPI == 1) v0 = gelu_exact(v0);
        const size_t off = (size_t)mg * N + ng;
        if (outf) outf[off] = v0;
        if (outb) outb[off] = f2bf(v0);
      }
    }
  }
}

// ---------------- flash attention (causal optional) ----------------
// q/k/v/ctx are bf16 in (B,S,D) layout, head slice = hh*64..+63.
// Block = 128 thr = 4 waves; wave handles 16 query rows; block-shared 32-key tiles.
// Softmax denominator accumulated as a WMMA row-sum (P @ ones) in C layout.
__global__ __launch_bounds__(128)
void attn_kernel(const u16* __restrict__ qg, const u16* __restrict__ kg,
                 const u16* __restrict__ vg, u16* __restrict__ ctx,
                 const int* __restrict__ look, int Sn) {
  __shared__ u16 Ks[32 * 72];      // 32 keys x 64 dh (row-major, stride 72 -> 144B rows)
  __shared__ u16 Vt[64 * 40];      // 64 dh x 32 keys (transposed, stride 40 -> 80B rows)
  __shared__ u16 Ps[4 * 16 * 40];  // per-wave P staging (C->A relayout)
  const int lane = threadIdx.x & 31;
  const int w    = threadIdx.x >> 5;
  const int l16  = lane & 15;
  const int hl   = lane >> 4;
  const int q0b  = blockIdx.x * 64;
  const int q0   = q0b + w * 16;
  const int bh   = blockIdx.y;
  const int b    = bh >> 4;                 // H = 16
  const int hh   = bh & 15;
  const bool  causal = (look[0] != 0);
  const float scale  = 0.125f;              // 1/sqrt(64)

  // Q fragments: A-matrix layout, rows q0..q0+15, dh split 0..31 / 32..63
  const size_t qrow = (size_t)(b * Sn + q0 + l16) * D_MODEL + hh * DHEAD;
  Frag16 qf0, qf1;
  qf0.q[0] = *(const float4*)(qg + qrow + hl * 8);
  qf0.q[1] = *(const float4*)(qg + qrow + hl * 8 + 16);
  qf1.q[0] = *(const float4*)(qg + qrow + 32 + hl * 8);
  qf1.q[1] = *(const float4*)(qg + qrow + 32 + hl * 8 + 16);

  // all-ones bf16 B fragment for WMMA row sums
  Frag16 onesf;
#pragma unroll
  for (int j = 0; j < 16; ++j) onesf.s[j] = 0x3F80;  // bf16 1.0

  float mrow[8];
  v8f o[4] = {{}, {}, {}, {}};
  v8f lacc = {};
#pragma unroll
  for (int r = 0; r < 8; ++r) mrow[r] = -1e30f;

  // block-uniform key loop; per-wave activity guard keeps barriers convergent
  const int kend_blk = causal ? (q0b + 64) : Sn;

  for (int k0 = 0; k0 < kend_blk; k0 += 32) {
    __syncthreads();   // previous tile fully consumed
#pragma unroll
    for (int i = 0; i < 2; ++i) {
      const int idx = threadIdx.x + i * 128;  // 0..255
      const int key = idx >> 3;               // 0..31
      const int dhg = (idx & 7) * 8;          // 0,8,..,56
      const size_t gbase = (size_t)(b * Sn + k0 + key) * D_MODEL + hh * DHEAD + dhg;
      // K tile row-major: async global->LDS
      async_copy16(kg + gbase, &Ks[key * 72 + dhg]);
      // V tile transposed (scalar LDS stores, vector global load)
      F4U16 vv;
      vv.f = *(const float4*)(vg + gbase);
#pragma unroll
      for (int j = 0; j < 8; ++j) Vt[(dhg + j) * 40 + key] = vv.s[j];
    }
    WAIT_ASYNC(0);
    __syncthreads();

    const bool active = (!causal) || (k0 <= q0 + 15);
    if (active) {
      // scores: S = Q @ K^T, two 16x16 key subtiles, contract over DH=64
      v8f s0 = {}, s1 = {};
      {
        Frag16 kf;
        kf.q[0] = *(const float4*)(&Ks[l16 * 72 + hl * 8]);
        kf.q[1] = *(const float4*)(&Ks[l16 * 72 + hl * 8 + 16]);
        s0 = __builtin_amdgcn_wmma_f32_16x16x32_bf16(false, qf0.v, false, kf.v,
                                                     (short)0, s0, false, false);
        kf.q[0] = *(const float4*)(&Ks[l16 * 72 + 32 + hl * 8]);
        kf.q[1] = *(const float4*)(&Ks[l16 * 72 + 32 + hl * 8 + 16]);
        s0 = __builtin_amdgcn_wmma_f32_16x16x32_bf16(false, qf1.v, false, kf.v,
                                                     (short)0, s0, false, false);
        kf.q[0] = *(const float4*)(&Ks[(l16 + 16) * 72 + hl * 8]);
        kf.q[1] = *(const float4*)(&Ks[(l16 + 16) * 72 + hl * 8 + 16]);
        s1 = __builtin_amdgcn_wmma_f32_16x16x32_bf16(false, qf0.v, false, kf.v,
                                                     (short)0, s1, false, false);
        kf.q[0] = *(const float4*)(&Ks[(l16 + 16) * 72 + 32 + hl * 8]);
        kf.q[1] = *(const float4*)(&Ks[(l16 + 16) * 72 + 32 + hl * 8 + 16]);
        s1 = __builtin_amdgcn_wmma_f32_16x16x32_bf16(false, qf1.v, false, kf.v,
                                                     (short)0, s1, false, false);
      }
      // online softmax; rows live in lane-halves; DPP xor-butterfly row max
#pragma unroll
      for (int r = 0; r < 8; ++r) {
        const int qi = q0 + r + 8 * hl;
        float a0 = s0[r] * scale;
        float a1 = s1[r] * scale;
        if (causal) {
          if (k0 + l16 > qi)      a0 += -1e9f;
          if (k0 + 16 + l16 > qi) a1 += -1e9f;
        }
        float mx = fmaxf(a0, a1);
        mx = XORMAX(mx, 1);
        mx = XORMAX(mx, 2);
        mx = XORMAX(mx, 4);
        mx = XORMAX(mx, 8);
        const float mnew = fmaxf(mrow[r], mx);
        const float corr = __expf(mrow[r] - mnew);
        const float p0 = __expf(a0 - mnew);
        const float p1 = __expf(a1 - mnew);
        mrow[r] = mnew;
        lacc[r] *= corr;
#pragma unroll
        for (int t = 0; t < 4; ++t) o[t][r] *= corr;
        Ps[w * 640 + (r + 8 * hl) * 40 + l16]      = f2bf(p0);
        Ps[w * 640 + (r + 8 * hl) * 40 + l16 + 16] = f2bf(p1);
      }
      asm volatile("s_wait_dscnt 0" ::: "memory");  // wave-private LDS relayout fence
      Frag16 pf;                                    // P as A-matrix (16x32 keys)
      pf.q[0] = *(const float4*)(&Ps[w * 640 + l16 * 40 + hl * 8]);
      pf.q[1] = *(const float4*)(&Ps[w * 640 + l16 * 40 + hl * 8 + 16]);
      // softmax denominator: lacc += P @ ones (row sums, same C layout as O)
      lacc = __builtin_amdgcn_wmma_f32_16x16x32_bf16(false, pf.v, false, onesf.v,
                                                     (short)0, lacc, false, false);
      // O += P @ V : four 16-wide dh tiles from transposed V staging
#pragma unroll
      for (int t = 0; t < 4; ++t) {
        Frag16 vf;
        vf.q[0] = *(const float4*)(&Vt[(t * 16 + l16) * 40 + hl * 8]);
        vf.q[1] = *(const float4*)(&Vt[(t * 16 + l16) * 40 + hl * 8 + 16]);
        o[t] = __builtin_amdgcn_wmma_f32_16x16x32_bf16(false, pf.v, false, vf.v,
                                                       (short)0, o[t], false, false);
      }
    }
  }
  // normalize and write ctx (bf16)
#pragma unroll
  for (int r = 0; r < 8; ++r) {
    const int   qi  = q0 + r + 8 * hl;
    const float inv = 1.0f / lacc[r];
    const size_t base = (size_t)(b * Sn + qi) * D_MODEL + hh * DHEAD + l16;
#pragma unroll
    for (int t = 0; t < 4; ++t) ctx[base + t * 16] = f2bf(o[t][r] * inv);
  }
}

// ---------------- residual add + layernorm, dual f32/bf16 output ----------------
__global__ __launch_bounds__(256)
void ln_residual(const float* __restrict__ x, const float* __restrict__ res,
                 const float* __restrict__ g, const float* __restrict__ be,
                 float* __restrict__ outf, u16* __restrict__ outb) {
  const int row = blockIdx.x;
  const size_t base = (size_t)row * D_MODEL;
  float v[4];
  float s = 0.f, sq = 0.f;
#pragma unroll
  for (int i = 0; i < 4; ++i) {
    const int idx = threadIdx.x + i * 256;
    const float t = x[base + idx] + res[base + idx];
    v[i] = t; s += t; sq += t * t;
  }
#pragma unroll
  for (int m = 16; m >= 1; m >>= 1) { s += __shfl_xor(s, m, 32); sq += __shfl_xor(sq, m, 32); }
  __shared__ float ss[8], sqs[8];
  const int wv = threadIdx.x >> 5;
  if ((threadIdx.x & 31) == 0) { ss[wv] = s; sqs[wv] = sq; }
  __syncthreads();
  s = 0.f; sq = 0.f;
#pragma unroll
  for (int i = 0; i < 8; ++i) { s += ss[i]; sq += sqs[i]; }
  const float mu   = s * (1.0f / D_MODEL);
  const float var  = sq * (1.0f / D_MODEL) - mu * mu;
  const float rstd = rsqrtf(var + LN_EPS);
#pragma unroll
  for (int i = 0; i < 4; ++i) {
    const int idx = threadIdx.x + i * 256;
    const float ov = (v[i] - mu) * rstd * g[idx] + be[idx];
    if (outf) outf[base + idx] = ov;
    if (outb) outb[base + idx] = f2bf(ov);
  }
}

// ---------------- host orchestration ----------------
extern "C" void kernel_launch(void* const* d_in, const int* in_sizes, int n_in,
                              void* d_out, int out_size, void* d_ws, size_t ws_size,
                              hipStream_t stream) {
  (void)in_sizes; (void)n_in; (void)out_size; (void)ws_size;
  const float* src  = (const float*)d_in[0];
  const int*   look = (const int*)d_in[1];
  const float* Wq = (const float*)d_in[2];  const float* bq  = (const float*)d_in[3];
  const float* Wk = (const float*)d_in[4];  const float* bk  = (const float*)d_in[5];
  const float* Wv = (const float*)d_in[6];  const float* bv  = (const float*)d_in[7];
  const float* Wo = (const float*)d_in[8];  const float* bo  = (const float*)d_in[9];
  const float* g1 = (const float*)d_in[10]; const float* be1 = (const float*)d_in[11];
  const float* W1 = (const float*)d_in[12]; const float* b1  = (const float*)d_in[13];
  const float* W2 = (const float*)d_in[14]; const float* b2  = (const float*)d_in[15];
  const float* g2 = (const float*)d_in[16]; const float* be2 = (const float*)d_in[17];
  float* out = (float*)d_out;

  const int Bn = 2, Sn = 2048, Dn = D_MODEL, Fn = F_FF;
  const int M = Bn * Sn;   // 4096 rows

  uint8_t* p = (uint8_t*)d_ws;
  auto alloc = [&](size_t bytes) -> void* {
    void* r = (void*)p;
    p += (bytes + 255) & ~(size_t)255;
    return r;
  };

  u16*   hb   = (u16*)alloc((size_t)M * Dn * 2);   // bf16 of current h
  u16*   qb   = (u16*)alloc((size_t)M * Dn * 2);
  u16*   kb   = (u16*)alloc((size_t)M * Dn * 2);
  u16*   vb   = (u16*)alloc((size_t)M * Dn * 2);
  u16*   ctxb = (u16*)alloc((size_t)M * Dn * 2);
  u16*   tb   = qb;  // FFN intermediate bf16 (M*F*2 = 32MB) reuses dead q/k/v/ctx region
  float* tmpf = (float*)alloc((size_t)M * Dn * 4); // attn-out / ffn-out f32
  float* h1   = (float*)alloc((size_t)M * Dn * 4); // post-attention LN (f32)
  u16*   h1b  = (u16*)alloc((size_t)M * Dn * 2);
  float* hcur = (float*)alloc((size_t)M * Dn * 4); // layer output f32
  u16* WqT = (u16*)alloc((size_t)Dn * Dn * 2);
  u16* WkT = (u16*)alloc((size_t)Dn * Dn * 2);
  u16* WvT = (u16*)alloc((size_t)Dn * Dn * 2);
  u16* WoT = (u16*)alloc((size_t)Dn * Dn * 2);
  u16* W1T = (u16*)alloc((size_t)Dn * Fn * 2);
  u16* W2T = (u16*)alloc((size_t)Fn * Dn * 2);

  cvt_f32_bf16<<<dim3((M * Dn + 255) / 256), 256, 0, stream>>>(src, hb, M * Dn);

  const dim3 gdd((Dn * Dn + 255) / 256);
  const dim3 gdf(((size_t)Dn * Fn + 255) / 256);
  const dim3 gproj(M / 128, Dn / 128);
  const dim3 gff1(M / 128, Fn / 128);
  const dim3 gattn(Sn / 64, Bn * N_HEAD);

  for (int l = 0; l < N_LAYER; ++l) {
    const float* hin = (l == 0) ? src : hcur;
    const size_t wdd = (size_t)l * Dn * Dn;
    const size_t wdf = (size_t)l * Dn * Fn;

    wcvt_t<<<gdd, 256, 0, stream>>>(Wq + wdd, WqT, Dn, Dn);
    wcvt_t<<<gdd, 256, 0, stream>>>(Wk + wdd, WkT, Dn, Dn);
    wcvt_t<<<gdd, 256, 0, stream>>>(Wv + wdd, WvT, Dn, Dn);
    wcvt_t<<<gdd, 256, 0, stream>>>(Wo + wdd, WoT, Dn, Dn);
    wcvt_t<<<gdf, 256, 0, stream>>>(W1 + wdf, W1T, Dn, Fn);
    wcvt_t<<<gdf, 256, 0, stream>>>(W2 + wdf, W2T, Fn, Dn);

    gemm_bf16<0><<<gproj, 256, 0, stream>>>(hb, WqT, bq + (size_t)l * Dn, nullptr, qb, M, Dn, Dn);
    gemm_bf16<0><<<gproj, 256, 0, stream>>>(hb, WkT, bk + (size_t)l * Dn, nullptr, kb, M, Dn, Dn);
    gemm_bf16<0><<<gproj, 256, 0, stream>>>(hb, WvT, bv + (size_t)l * Dn, nullptr, vb, M, Dn, Dn);

    attn_kernel<<<gattn, 128, 0, stream>>>(qb, kb, vb, ctxb, look, Sn);

    gemm_bf16<0><<<gproj, 256, 0, stream>>>(ctxb, WoT, bo + (size_t)l * Dn, tmpf, nullptr, M, Dn, Dn);
    ln_residual<<<M, 256, 0, stream>>>(tmpf, hin, g1 + (size_t)l * Dn, be1 + (size_t)l * Dn, h1, h1b);

    gemm_bf16<1><<<gff1, 256, 0, stream>>>(h1b, W1T, b1 + (size_t)l * Fn, nullptr, tb, M, Fn, Dn);
    gemm_bf16<0><<<gproj, 256, 0, stream>>>(tb, W2T, b2 + (size_t)l * Dn, tmpf, nullptr, M, Dn, Fn);

    float* hnext = (l == N_LAYER - 1) ? out : hcur;
    ln_residual<<<M, 256, 0, stream>>>(tmpf, h1, g2 + (size_t)l * Dn, be2 + (size_t)l * Dn, hnext, hb);
  }
}
